// Net_75857712382022
// MI455X (gfx1250) — compile-verified, matrix operand order checked by reading, not run on hardware
//
#include <hip/hip_runtime.h>
#include <hip/hip_bf16.h>

typedef __attribute__((ext_vector_type(16))) __bf16 v16bf;
typedef __attribute__((ext_vector_type(8)))  __bf16 bf16x8;
typedef __attribute__((ext_vector_type(8)))  float  v8f;
typedef __attribute__((ext_vector_type(4)))  float  f32x4;

#define GEMM_BM 64
#define GEMM_BN 64
#define GEMM_BK 32
#define LDSK 40   // bf16 elems per LDS row (80B row -> every 8-elem chunk 16B aligned)

// ---------------------------------------------------------------------------
// WMMA GEMM: C[M,N] = A[M,K] @ W[K,N] (+bias)(+relu).
//  - A: f32, row stride lda (multiple of 4, >= Kpad), loaded as 2x b128 + cvt
//  - Wt: pre-converted bf16, transposed [N][Kpad], zero-padded K tail;
//        staged with global_load_async_to_lds_b128 (ASYNCcnt path)
//  - 256 thr = 8 waves; 64x64 block tile; 2 wmma per wave per K-step
// ---------------------------------------------------------------------------
__global__ __launch_bounds__(256)
void gemm_wmma_bf16(const float* __restrict__ A, int lda,
                    const __bf16* __restrict__ Wt,
                    const float* __restrict__ bias, float* __restrict__ C,
                    int M, int N, int Kpad, int relu) {
  __shared__ __attribute__((aligned(16))) __bf16 sA[GEMM_BM * LDSK];
  __shared__ __attribute__((aligned(16))) __bf16 sB[GEMM_BN * LDSK];
  const int m0 = blockIdx.y * GEMM_BM;
  const int n0 = blockIdx.x * GEMM_BN;
  const int tid = threadIdx.x;
  const int wave = tid >> 5, lane = tid & 31;
  const int wm = (wave >> 1) * 16;     // 0,16,32,48
  const int wn = (wave & 1) * 32;      // 0,32 (two 16-wide tiles each)
  const int row = lane & 15;
  const int kb  = (lane >> 4) * 8;     // K base 0 / 8 per ISA 7.12.2

  const int ar = tid >> 2;             // A stage: row 0..63
  const int ak = (tid & 3) * 8;        // A stage: k 0,8,16,24
  const int br = tid >> 2;             // B stage: row 0..63
  const int bk = (tid & 3) * 8;

  const float*  Arow = A  + (size_t)min(m0 + ar, M - 1) * lda;
  const __bf16* Brow = Wt + (size_t)min(n0 + br, N - 1) * Kpad;
  const unsigned sBoff = (unsigned)(uintptr_t)&sB[br * LDSK + bk];

  v8f acc0 = {}, acc1 = {};

  for (int k0 = 0; k0 < Kpad; k0 += GEMM_BK) {
    // B tile: async bf16 copy global -> LDS (one b128 per lane)
    {
      const __bf16* g = Brow + k0 + bk;
      asm volatile("global_load_async_to_lds_b128 %0, %1, off"
                   :: "v"(sBoff), "v"(g) : "memory");
    }
    // A tile: unguarded vector loads + packed bf16 convert + b128 LDS store
    {
      const f32x4* pa = (const f32x4*)(Arow + k0 + ak);
      f32x4 lo = pa[0], hi = pa[1];
      bf16x8 av;
#pragma unroll
      for (int j = 0; j < 4; ++j) { av[j] = (__bf16)lo[j]; av[j + 4] = (__bf16)hi[j]; }
      *(bf16x8*)&sA[ar * LDSK + ak] = av;
    }
    if (k0 + GEMM_BK < Kpad)
      __builtin_prefetch(Arow + k0 + GEMM_BK + ak, 0, 1);  // global_prefetch_b8
    asm volatile("s_wait_asynccnt 0x0" ::: "memory");
    __syncthreads();

    // Fragments: two aligned b128 LDS loads each
    bf16x8 a0  = *(const bf16x8*)&sA[(wm + row) * LDSK + kb];
    bf16x8 a1  = *(const bf16x8*)&sA[(wm + row) * LDSK + kb + 16];
    bf16x8 b00 = *(const bf16x8*)&sB[(wn + row) * LDSK + kb];
    bf16x8 b01 = *(const bf16x8*)&sB[(wn + row) * LDSK + kb + 16];
    bf16x8 b10 = *(const bf16x8*)&sB[(wn + 16 + row) * LDSK + kb];
    bf16x8 b11 = *(const bf16x8*)&sB[(wn + 16 + row) * LDSK + kb + 16];
    v16bf af, bf0, bf1;
#pragma unroll
    for (int j = 0; j < 8; ++j) {
      af[j]  = a0[j];  af[j + 8]  = a1[j];
      bf0[j] = b00[j]; bf0[j + 8] = b01[j];
      bf1[j] = b10[j]; bf1[j + 8] = b11[j];
    }
    acc0 = __builtin_amdgcn_wmma_f32_16x16x32_bf16(false, af, false, bf0,
                                                   (short)0, acc0, false, false);
    acc1 = __builtin_amdgcn_wmma_f32_16x16x32_bf16(false, af, false, bf1,
                                                   (short)0, acc1, false, false);
    __syncthreads();
  }

  // Store per ISA C/D layout: lanes 0-15 -> M=r, lanes 16-31 -> M=r+8
  const int rbase = m0 + wm + ((lane >> 4) * 8);
#pragma unroll
  for (int t = 0; t < 2; ++t) {
    int col = n0 + wn + t * 16 + (lane & 15);
    if (col < N) {
      float bi = bias ? bias[col] : 0.f;
      v8f acc = t ? acc1 : acc0;
#pragma unroll
      for (int r = 0; r < 8; ++r) {
        int rr = rbase + r;
        if (rr < M) {
          float v = acc[r] + bi;
          if (relu) v = fmaxf(v, 0.f);
          C[(size_t)rr * N + col] = v;
        }
      }
    }
  }
}

// ---------------------------------------------------------------------------
// One-shot converters (run each launch; deterministic)
// ---------------------------------------------------------------------------
__global__ void convert_wt(const float* __restrict__ W, __bf16* __restrict__ Wt,
                           int K, int N, int Kpad) {
  int i = blockIdx.x * blockDim.x + threadIdx.x;
  if (i >= N * Kpad) return;
  int n = i / Kpad, k = i - n * Kpad;
  Wt[i] = (k < K) ? (__bf16)W[(size_t)k * N + n] : (__bf16)0.f;
}

__global__ void pad_copy(const float* __restrict__ x, float* __restrict__ xp,
                         int n, int K, int Kpad) {
  int i = blockIdx.x * blockDim.x + threadIdx.x;
  if (i >= n * Kpad) return;
  int r = i / Kpad, k = i - r * Kpad;
  xp[i] = (k < K) ? x[(size_t)r * K + k] : 0.f;
}

// ---------------------------------------------------------------------------
// Helper / edge kernels
// ---------------------------------------------------------------------------
__device__ inline void atomicMaxF(float* addr, float val) {
  unsigned int* ua = (unsigned int*)addr;
  unsigned int old = *ua;
  while (__uint_as_float(old) < val) {
    unsigned int assumed = old;
    old = atomicCAS(ua, assumed, __float_as_uint(val));
    if (old == assumed) break;
  }
}

__global__ void fill_f32(float* p, float v, int n) {
  int i = blockIdx.x * blockDim.x + threadIdx.x;
  if (i < n) p[i] = v;
}

__global__ void deg_kernel(const int* __restrict__ dst, const float* __restrict__ w,
                           float* __restrict__ deg, int E) {
  int e = blockIdx.x * blockDim.x + threadIdx.x;
  if (e < E) atomicAdd(&deg[dst[e]], w[e]);
}

__global__ void dinv_kernel(float* __restrict__ deg, float* __restrict__ dinv, int n) {
  int i = blockIdx.x * blockDim.x + threadIdx.x;
  if (i < n) { float d = deg[i] + 1.f; deg[i] = d; dinv[i] = rsqrtf(d); }
}

__global__ void norm_kernel(const int* __restrict__ src, const int* __restrict__ dst,
                            const float* __restrict__ w, const float* __restrict__ dinv,
                            float* __restrict__ nrm, int E) {
  int e = blockIdx.x * blockDim.x + threadIdx.x;
  if (e < E) nrm[e] = dinv[src[e]] * w[e] * dinv[dst[e]];
}

// out[dst,:] += h[src,:] * nrm[e]   (one block per edge; ld = row stride)
__global__ void scatter_scaled(const float* __restrict__ h, const int* __restrict__ src,
                               const int* __restrict__ dst, const float* __restrict__ nrm,
                               float* __restrict__ out, int E, int dim, int ld) {
  int e = blockIdx.x;
  int s = src[e], d = dst[e];
  float c = nrm[e];
  for (int f = threadIdx.x; f < dim; f += blockDim.x)
    atomicAdd(&out[(size_t)d * ld + f], h[(size_t)s * ld + f] * c);
}

// out[dst,:] += h[src,:]   (one block per edge, GIN)
__global__ void scatter_plain(const float* __restrict__ h, const int* __restrict__ src,
                              const int* __restrict__ dst, float* __restrict__ out,
                              int E, int dim, int ld) {
  int e = blockIdx.x;
  int s = src[e], d = dst[e];
  for (int f = threadIdx.x; f < dim; f += blockDim.x)
    atomicAdd(&out[(size_t)d * ld + f], h[(size_t)s * ld + f]);
}

__global__ void add_inplace(float* __restrict__ out, const float* __restrict__ h, int n) {
  int i = blockIdx.x * blockDim.x + threadIdx.x;
  if (i < n) out[i] += h[i];
}

// out = BN(relu(agg + (hw?hw/deg:0) + (bias?bias:0)))  (eval BN, eps=1e-5)
__global__ void finalize_bn(const float* __restrict__ agg, const float* __restrict__ hw,
                            const float* __restrict__ deg, const float* __restrict__ bias,
                            const float* __restrict__ g, const float* __restrict__ bb,
                            const float* __restrict__ rm, const float* __restrict__ rv,
                            float* __restrict__ out, int n, int dim) {
  int i = blockIdx.x * blockDim.x + threadIdx.x;
  if (i >= n * dim) return;
  int f = i % dim, node = i / dim;
  float v = agg[i];
  if (hw)   v += hw[i] / deg[node];
  if (bias) v += bias[f];
  v = fmaxf(v, 0.f);
  v = (v - rm[f]) * rsqrtf(rv[f] + 1e-5f) * g[f] + bb[f];
  out[i] = v;
}

__global__ void gat_heads(const float* __restrict__ xl, const float* __restrict__ as,
                          const float* __restrict__ ad, float* __restrict__ asum,
                          float* __restrict__ adsum, int n, int H, int C) {
  int i = blockIdx.x * blockDim.x + threadIdx.x;
  if (i >= n * H) return;
  int node = i / H, h = i - node * H;
  const float* xp = xl + (size_t)node * H * C + h * C;
  const float* ap = as + h * C;
  const float* dp = ad + h * C;
  float ss = 0.f, sd = 0.f;
  for (int c = 0; c < C; ++c) { float v = xp[c]; ss += v * ap[c]; sd += v * dp[c]; }
  asum[i] = ss; adsum[i] = sd;
}

__global__ void gat_pass1(const float* __restrict__ asum, const float* __restrict__ adsum,
                          const int* __restrict__ src, const int* __restrict__ dst,
                          float* __restrict__ ebuf, float* __restrict__ m,
                          int E, int n, int H) {
  int i = blockIdx.x * blockDim.x + threadIdx.x;
  if (i >= (E + n) * H) return;
  int eid = i / H, h = i - eid * H;
  int s = eid < E ? src[eid] : eid - E;
  int d = eid < E ? dst[eid] : eid - E;
  float e = asum[s * H + h] + adsum[d * H + h];
  e = e > 0.f ? e : 0.2f * e;
  ebuf[i] = e;
  atomicMaxF(&m[d * H + h], e);
}

__global__ void gat_pass2(float* __restrict__ ebuf, const float* __restrict__ m,
                          float* __restrict__ den, const int* __restrict__ dst,
                          int E, int n, int H) {
  int i = blockIdx.x * blockDim.x + threadIdx.x;
  if (i >= (E + n) * H) return;
  int eid = i / H, h = i - eid * H;
  int d = eid < E ? dst[eid] : eid - E;
  float ex = __expf(ebuf[i] - m[d * H + h]);
  ebuf[i] = ex;
  atomicAdd(&den[d * H + h], ex);
}

__global__ void gat_scatter(const float* __restrict__ xl, const float* __restrict__ ebuf,
                            const float* __restrict__ den, const int* __restrict__ src,
                            const int* __restrict__ dst, float* __restrict__ out,
                            int E, int n, int H, int C) {
  int eid = blockIdx.x;
  int s = eid < E ? src[eid] : eid - E;
  int d = eid < E ? dst[eid] : eid - E;
  __shared__ float coef[8];
  if ((int)threadIdx.x < H)
    coef[threadIdx.x] = ebuf[eid * H + threadIdx.x] / (den[d * H + threadIdx.x] + 1e-16f);
  __syncthreads();
  int HC = H * C;
  for (int f = threadIdx.x; f < HC; f += blockDim.x) {
    int h = f / C;
    atomicAdd(&out[(size_t)d * HC + f], xl[(size_t)s * HC + f] * coef[h]);
  }
}

__global__ void pool_kernel(const float* __restrict__ h, const int* __restrict__ batch,
                            float* __restrict__ pooled, int n, int dim, int coloff,
                            int stride) {
  int node = blockIdx.x;
  int b = batch[node];
  for (int f = threadIdx.x; f < dim; f += blockDim.x)
    atomicAdd(&pooled[(size_t)b * stride + coloff + f], h[(size_t)node * dim + f]);
}

// ---------------------------------------------------------------------------
// Host side
// ---------------------------------------------------------------------------
extern "C" void kernel_launch(void* const* d_in, const int* in_sizes, int n_in,
                              void* d_out, int out_size, void* d_ws, size_t ws_size,
                              hipStream_t stream) {
  (void)in_sizes; (void)n_in; (void)out_size; (void)ws_size;
  const int N = 10000, E = 160000, F_IN = 373, G = 64, H = 3;
  const int FPAD = 384;  // F_IN rounded to 32

  const float* x   = (const float*)d_in[0];
  const int* ei    = (const int*)d_in[1];
  const int* srcp  = ei;
  const int* dstp  = ei + E;
  const float* ew  = (const float*)d_in[2];
  const int* batch = (const int*)d_in[3];
  auto P = [&](int i) { return (const float*)d_in[i]; };

  char* base = (char*)d_ws;
  size_t off = 0;
  auto allocB = [&](size_t bytes) -> char* {
    char* p = base + off;
    off += (bytes + 255) & ~(size_t)255;
    return p;
  };
  auto alloc = [&](size_t nf) -> float* { return (float*)allocB(nf * 4); };

  float* deg    = alloc(N);
  float* dinv   = alloc(N);
  float* nrm    = alloc(E);
  float* asum   = alloc((size_t)N * H);
  float* adsum  = alloc((size_t)N * H);
  float* mbuf   = alloc((size_t)N * H);
  float* den    = alloc((size_t)N * H);
  float* ebuf   = alloc((size_t)(E + N) * H);
  float* xpad   = alloc((size_t)N * FPAD);
  float* bufA   = alloc((size_t)N * 768);
  float* bufB   = alloc((size_t)N * 768);
  float* bufC   = alloc((size_t)N * 768);
  float* pooled = alloc((size_t)G * 576);
  float* fc1o   = alloc((size_t)G * 256);
  float* fc2o   = alloc((size_t)G * 64);

  auto KPAD = [](int k) { return (k + 31) & ~31; };
  auto prepW = [&](int idx, int K, int Nn) -> const __bf16* {
    int kp = KPAD(K);
    __bf16* wt = (__bf16*)allocB((size_t)Nn * kp * sizeof(__bf16));
    int total = Nn * kp;
    convert_wt<<<(total + 255) / 256, 256, 0, stream>>>(P(idx), wt, K, Nn, kp);
    return wt;
  };
  const __bf16* Wgcn1  = prepW(4, 373, 256);
  const __bf16* Wgcn2  = prepW(10, 256, 128);
  const __bf16* Wgcn3  = prepW(16, 128, 128);
  const __bf16* Wgat1  = prepW(22, 373, 768);
  const __bf16* Wgat2  = prepW(30, 768, 384);
  const __bf16* Wgat3  = prepW(38, 384, 384);
  const __bf16* Wgin1a = prepW(46, 373, 256);
  const __bf16* Wgin1b = prepW(48, 256, 256);
  const __bf16* Wgin2a = prepW(54, 256, 128);
  const __bf16* Wgin2b = prepW(56, 128, 128);
  const __bf16* Wgin3a = prepW(62, 128, 64);
  const __bf16* Wgin3b = prepW(64, 64, 64);
  const __bf16* Wfc1   = prepW(70, 576, 256);
  const __bf16* Wfc2   = prepW(72, 256, 64);
  const __bf16* Wfc3   = prepW(74, 64, 1);

  pad_copy<<<(N * FPAD + 255) / 256, 256, 0, stream>>>(x, xpad, N, F_IN, FPAD);

  auto gemm = [&](const float* A, int lda, const __bf16* Wt, const float* b, float* C,
                  int M, int Nn, int K, int relu) {
    dim3 grid((Nn + GEMM_BN - 1) / GEMM_BN, (M + GEMM_BM - 1) / GEMM_BM);
    gemm_wmma_bf16<<<grid, 256, 0, stream>>>(A, lda, Wt, b, C, M, Nn, KPAD(K), relu);
  };

  // ---- GCN normalization ----
  hipMemsetAsync(deg, 0, (size_t)N * 4, stream);
  deg_kernel<<<(E + 255) / 256, 256, 0, stream>>>(dstp, ew, deg, E);
  dinv_kernel<<<(N + 255) / 256, 256, 0, stream>>>(deg, dinv, N);
  norm_kernel<<<(E + 255) / 256, 256, 0, stream>>>(srcp, dstp, ew, dinv, nrm, E);
  hipMemsetAsync(pooled, 0, (size_t)G * 576 * 4, stream);

  // ---- GCN branch ----
  auto gcn_layer = [&](const float* h, int lda, int din, int dout,
                       const __bf16* Wt, int bi, int bni) {
    gemm(h, lda, Wt, nullptr, bufA, N, dout, din, 0);
    hipMemsetAsync(bufB, 0, (size_t)N * dout * 4, stream);
    scatter_scaled<<<dim3(E), dim3(256), 0, stream>>>(bufA, srcp, dstp, nrm, bufB,
                                                      E, dout, dout);
    int total = N * dout;
    finalize_bn<<<(total + 255) / 256, 256, 0, stream>>>(
        bufB, bufA, deg, P(bi), P(bni), P(bni + 1), P(bni + 2), P(bni + 3),
        bufB, N, dout);
  };
  gcn_layer(xpad, FPAD, 373, 256, Wgcn1, 5, 6);
  gcn_layer(bufB, 256,  256, 128, Wgcn2, 11, 12);
  gcn_layer(bufB, 128,  128, 128, Wgcn3, 17, 18);
  pool_kernel<<<dim3(N), dim3(256), 0, stream>>>(bufB, batch, pooled, N, 128, 0, 576);

  // ---- GAT branch ----
  auto gat_layer = [&](const float* h, int lda, int din, int C, const __bf16* Wt,
                       int asi, int adi, int bi, int bni) {
    int HC = H * C;
    gemm(h, lda, Wt, nullptr, bufA, N, HC, din, 0);
    gat_heads<<<(N * H + 255) / 256, 256, 0, stream>>>(bufA, P(asi), P(adi),
                                                       asum, adsum, N, H, C);
    fill_f32<<<(N * H + 255) / 256, 256, 0, stream>>>(mbuf, -3.0e38f, N * H);
    hipMemsetAsync(den, 0, (size_t)N * H * 4, stream);
    int EL = E + N, tt = EL * H;
    gat_pass1<<<(tt + 255) / 256, 256, 0, stream>>>(asum, adsum, srcp, dstp,
                                                    ebuf, mbuf, E, N, H);
    gat_pass2<<<(tt + 255) / 256, 256, 0, stream>>>(ebuf, mbuf, den, dstp, E, N, H);
    hipMemsetAsync(bufB, 0, (size_t)N * HC * 4, stream);
    gat_scatter<<<dim3(EL), dim3(256), 0, stream>>>(bufA, ebuf, den, srcp, dstp,
                                                    bufB, E, N, H, C);
    int total = N * HC;
    finalize_bn<<<(total + 255) / 256, 256, 0, stream>>>(
        bufB, nullptr, nullptr, P(bi), P(bni), P(bni + 1), P(bni + 2), P(bni + 3),
        bufB, N, HC);
  };
  gat_layer(xpad, FPAD, 373, 256, Wgat1, 23, 24, 25, 26);
  gat_layer(bufB, 768,  768, 128, Wgat2, 31, 32, 33, 34);
  gat_layer(bufB, 384,  384, 128, Wgat3, 39, 40, 41, 42);
  pool_kernel<<<dim3(N), dim3(256), 0, stream>>>(bufB, batch, pooled, N, 384, 128, 576);

  // ---- GIN branch ----
  auto gin_layer = [&](const float* h, int ld, int din, int dmid,
                       const __bf16* W1, int b1, const __bf16* W2, int b2, int bni) {
    hipMemsetAsync(bufA, 0, (size_t)N * ld * 4, stream);
    scatter_plain<<<dim3(E), dim3(256), 0, stream>>>(h, srcp, dstp, bufA, E, din, ld);
    add_inplace<<<(N * ld + 255) / 256, 256, 0, stream>>>(bufA, h, N * ld);
    gemm(bufA, ld, W1, P(b1), bufB, N, dmid, din, 1);
    gemm(bufB, dmid, W2, P(b2), bufC, N, dmid, dmid, 0);
    int total = N * dmid;
    finalize_bn<<<(total + 255) / 256, 256, 0, stream>>>(
        bufC, nullptr, nullptr, nullptr, P(bni), P(bni + 1), P(bni + 2), P(bni + 3),
        bufC, N, dmid);
  };
  gin_layer(xpad, FPAD, 373, 256, Wgin1a, 47, Wgin1b, 49, 50);
  gin_layer(bufC, 256,  256, 128, Wgin2a, 55, Wgin2b, 57, 58);
  gin_layer(bufC, 128,  128, 64,  Wgin3a, 63, Wgin3b, 65, 66);
  pool_kernel<<<dim3(N), dim3(256), 0, stream>>>(bufC, batch, pooled, N, 64, 512, 576);

  // ---- FC head ----
  gemm(pooled, 576, Wfc1, P(71), fc1o, G, 256, 576, 1);
  gemm(fc1o,   256, Wfc2, P(73), fc2o, G, 64,  256, 1);
  gemm(fc2o,   64,  Wfc3, P(75), (float*)d_out, G, 1, 64, 1);  // final relu fused
}
